// SparseSegmentation_24799141167287
// MI455X (gfx1250) — matrix operand order repaired and checked by reading, not compiled, not run on hardware
//
#include <hip/hip_runtime.h>

// ---------------------------------------------------------------------------
// Sparse U-Net segmentation forward for MI455X (gfx1250, wave32, WMMA).
// 3x3 submanifold convs = gather-GEMMs on v_wmma_f32_16x16x32_f16 (fp32 acc).
// Weight tiles staged into LDS once per block via the Tensor Data Mover
// (tensor_load_to_lds + s_wait_tensorcnt) and consumed via ds_load_b128.
// ---------------------------------------------------------------------------

typedef _Float16 f16;
typedef f16          v16h   __attribute__((ext_vector_type(16)));
typedef float        v8f    __attribute__((ext_vector_type(8)));
typedef unsigned int uint4v __attribute__((ext_vector_type(4)));
typedef unsigned int u32x4  __attribute__((ext_vector_type(4)));
typedef int          i32x4  __attribute__((ext_vector_type(4)));
typedef int          i32x8  __attribute__((ext_vector_type(8)));

#if defined(__has_builtin)
#  if __has_builtin(__builtin_amdgcn_tensor_load_to_lds)
#    define HAVE_TDM 1
#  endif
#endif

#define BN_EPS 1e-4f

static __device__ __forceinline__ float h2f(unsigned short bits) {
    union { f16 h; unsigned short b; } cv; cv.b = bits; return (float)cv.h;
}
static __device__ __forceinline__ unsigned short f2h(float v) {
    union { f16 h; unsigned short b; } cv; cv.h = (f16)v; return cv.b;
}

// ---------------------------------------------------------------------------
// memzero
// ---------------------------------------------------------------------------
__global__ void k_zero_f32(float* __restrict__ p, long n) {
    long i = (long)blockIdx.x * blockDim.x + threadIdx.x;
    if (i < n) p[i] = 0.0f;
}

// ---------------------------------------------------------------------------
// input 3x3 submanifold conv, Cin=1 -> Cout=32 (too thin for WMMA; VALU)
// ---------------------------------------------------------------------------
__global__ void k_win(const float* __restrict__ feats, const int* __restrict__ nbr,
                      const float* __restrict__ w, float* __restrict__ out, int N) {
    long t = (long)blockIdx.x * blockDim.x + threadIdx.x;
    if (t >= (long)N * 32) return;
    int n = (int)(t >> 5), d = (int)(t & 31);
    float acc = 0.0f;
#pragma unroll
    for (int k = 0; k < 9; ++k) {
        int s = nbr[n * 9 + k];
        if (s >= 0) acc += feats[s] * w[k * 32 + d];
    }
    out[(long)n * 32 + d] = acc;
}

// ---------------------------------------------------------------------------
// BatchNorm statistics: one block per channel; stats[c]=mean, stats[C+c]=rstd
// ---------------------------------------------------------------------------
__global__ void k_bn_stats(const float* __restrict__ x, float* __restrict__ stats,
                           int N, int C) {
    int c = blockIdx.x;
    float s = 0.0f, q = 0.0f;
    for (int i = threadIdx.x; i < N; i += blockDim.x) {
        float v = x[(long)i * C + c];
        s += v; q += v * v;
    }
    __shared__ float ss[256], sq[256];
    ss[threadIdx.x] = s; sq[threadIdx.x] = q;
    __syncthreads();
    for (int st = 128; st > 0; st >>= 1) {
        if (threadIdx.x < st) { ss[threadIdx.x] += ss[threadIdx.x + st];
                                sq[threadIdx.x] += sq[threadIdx.x + st]; }
        __syncthreads();
    }
    if (threadIdx.x == 0) {
        float m = ss[0] / (float)N;
        float v = sq[0] / (float)N - m * m;
        stats[c]     = m;
        stats[C + c] = rsqrtf(v + BN_EPS);
    }
}

// ---------------------------------------------------------------------------
// BN + ReLU, fp32 -> f16 staging for WMMA consumers
// ---------------------------------------------------------------------------
__global__ void k_bn_apply_h(const float* __restrict__ x, const float* __restrict__ stats,
                             const float* __restrict__ g, const float* __restrict__ b,
                             unsigned short* __restrict__ y, long total, int C) {
    long i = (long)blockIdx.x * blockDim.x + threadIdx.x;
    if (i >= total) return;
    int c = (int)(i % C);
    float v = (x[i] - stats[c]) * stats[C + c] * g[c] + b[c];
    y[i] = f2h(v > 0.0f ? v : 0.0f);
}

// ---------------------------------------------------------------------------
// convert smconv weights [K][Cin][Cout] fp32 -> transposed [K][Cout][Cin] f16
// ---------------------------------------------------------------------------
__global__ void k_cvt_wt(const float* __restrict__ w, unsigned short* __restrict__ wt,
                         int K, int Cin, int Cout) {
    long t = (long)blockIdx.x * blockDim.x + threadIdx.x;
    long tot = (long)K * Cin * Cout;
    if (t >= tot) return;
    int k   = (int)(t / ((long)Cin * Cout));
    long r  = t % ((long)Cin * Cout);
    int ci  = (int)(r / Cout);
    int co  = (int)(r % Cout);
    wt[((long)k * Cout + co) * Cin + ci] = f2h(w[t]);
}

// ---------------------------------------------------------------------------
// 3x3 submanifold conv as gather-GEMM on v_wmma_f32_16x16x32_f16.
//   xb : [N, Cin]  f16 (post BN+ReLU)       nbr: [N, 9] (-1 = inactive)
//   wt : [9, Cout, Cin] f16 (transposed)    out: [N, Cout] fp32
// Block = 128 threads = 4 waves; wave w handles row-tile 4*blockIdx.x + w;
// all waves share one 16-column weight tile staged in LDS ([9][16][Cin] f16),
// loaded by the Tensor Data Mover (3D tile: x=Cin, y=16 cols, z=9 taps).
// ---------------------------------------------------------------------------
__global__ __launch_bounds__(128) void k_smconv_wmma(
    const unsigned short* __restrict__ xb, const int* __restrict__ nbr,
    const unsigned short* __restrict__ wt, float* __restrict__ out,
    int N, int Cin, int Cout) {

    extern __shared__ unsigned short lds_w[];   // [9][16][Cin]
    const int tileElems = 16 * Cin;

    const int lane = threadIdx.x & 31;
    const int wave = threadIdx.x >> 5;
    const int row0 = (blockIdx.x * 4 + wave) * 16;
    const int col0 = blockIdx.y * 16;

#ifdef HAVE_TDM
    if (threadIdx.x < 32) {
        // D# group 0: count=1, lds_addr=0, 57-bit global tile address, type=2
        unsigned long long gaddr =
            (unsigned long long)(const void*)(wt + (unsigned long long)col0 * Cin);
        u32x4 g0 = { 1u, 0u,
                     (unsigned)(gaddr & 0xffffffffull),
                     (unsigned)((gaddr >> 32) & 0x1ffffffull) | (2u << 30) };
        // D# group 1: data_size=2B; tensor dims (Cin, Cout); tile (Cin,16,9);
        // strides: dim0_stride=Cin (y step), dim1_stride=Cout*Cin (z step)
        unsigned dim0 = (unsigned)Cin, dim1 = (unsigned)Cout;
        unsigned long long s0 = (unsigned long long)Cin;
        unsigned long long s1 = (unsigned long long)Cout * (unsigned)Cin;
        i32x8 g1;
        g1[0] = (int)(1u << 16);                                      // data_size=2B
        g1[1] = (int)((dim0 & 0xffffu) << 16);                        // tensor_dim0 lo
        g1[2] = (int)((dim0 >> 16) | ((dim1 & 0xffffu) << 16));       // dim0 hi|dim1 lo
        g1[3] = (int)((dim1 >> 16) | ((unsigned)Cin << 16));          // dim1 hi|tile_dim0
        g1[4] = (int)(16u | (9u << 16));                              // tile_dim1|tile_dim2
        g1[5] = (int)(s0 & 0xffffffffull);                            // dim0_stride lo
        g1[6] = (int)(((s0 >> 32) & 0xffffull) | ((s1 & 0xffffull) << 16));
        g1[7] = (int)((s1 >> 16) & 0xffffffffull);                    // dim1_stride hi
        i32x4 g2 = { 9, 0, 0, 0 };                                    // tensor_dim2=9
        i32x4 g3 = { 0, 0, 0, 0 };
#if __clang_major__ >= 23
        i32x8 g4 = { 0, 0, 0, 0, 0, 0, 0, 0 };
        __builtin_amdgcn_tensor_load_to_lds(g0, g1, g2, g3, g4, 0);
#else
        __builtin_amdgcn_tensor_load_to_lds(g0, g1, g2, g3, 0);
#endif
        __builtin_amdgcn_s_wait_tensorcnt(0);
    }
    __syncthreads();
#else
    // cooperative fallback: copy the tile through VGPRs
    for (int k = 0; k < 9; ++k)
        for (int e = threadIdx.x * 8; e < tileElems; e += 128 * 8)
            *(uint4v*)(lds_w + k * tileElems + e) =
                *(const uint4v*)(wt + ((unsigned long long)k * Cout + col0) * Cin + e);
    __syncthreads();
#endif

    // A (16-bit 16x32): lanes 0-15 = M, lane covers K in {kA..kA+7, kA+16..kA+23}
    const int m  = lane & 15;
    const int kA = (lane >> 4) * 8;
    // B (16-bit 32x16): lanes 0-15 = N, lane covers K in {kB..kB+7, kB+16..kB+23}
    const int n  = lane & 15;
    const int kB = (lane >> 4) * 8;

    const int  row  = row0 + m;
    const bool rowv = (row < N);

    // preload the 9 rulebook indices for this row
    int nb[9];
#pragma unroll
    for (int k = 0; k < 9; ++k) nb[k] = rowv ? nbr[row * 9 + k] : -1;

    v8f acc = {};

#pragma unroll 1
    for (int k = 0; k < 9; ++k) {
        const int  src   = nb[k];
        const bool amask = (src >= 0);
        const unsigned short* arow = xb + (long)(amask ? src : 0) * Cin;
        const unsigned short* lwp  = lds_w + (k * 16 + n) * Cin;

        if (k + 1 < 9 && nb[k + 1] >= 0)   // hide gather latency of next tap
            __builtin_prefetch(xb + (long)nb[k + 1] * Cin + kA, 0, 0);

#pragma unroll 2
        for (int c0 = 0; c0 < Cin; c0 += 32) {
            union { v16h v; uint4v q[2]; } A;
            uint4v z = {0u, 0u, 0u, 0u};
            A.q[0] = z; A.q[1] = z;
            if (amask) {
                A.q[0] = *(const uint4v*)(arow + c0 + kA);
                A.q[1] = *(const uint4v*)(arow + c0 + kA + 16);
            }
            union { v16h v; uint4v q[2]; } B;
            B.q[0] = *(const uint4v*)(lwp + c0 + kB);       // ds_load_b128
            B.q[1] = *(const uint4v*)(lwp + c0 + kB + 16);

            acc = __builtin_amdgcn_wmma_f32_16x16x32_f16(
                false, A.v, false, B.v, (short)0, acc, false, false);
        }
    }

    // C/D layout: lane holds column n, rows 8*(lane>>4) + g for g=0..7
    const int colw  = col0 + n;
    const int mbase = row0 + 8 * (lane >> 4);
#pragma unroll
    for (int g = 0; g < 8; ++g) {
        int r = mbase + g;
        if (r < N && colw < Cout) out[(long)r * Cout + colw] = acc[g];
    }
}

// ---------------------------------------------------------------------------
// stride-2 down conv: per fine point, y = x @ W[kidx], scatter-add to parent
// ---------------------------------------------------------------------------
__global__ void k_down(const unsigned short* __restrict__ xh, const int* __restrict__ inv,
                       const int* __restrict__ kidx, const float* __restrict__ w,
                       float* __restrict__ out, int N, int Cin, int Cout) {
    long t = (long)blockIdx.x * blockDim.x + threadIdx.x;
    if (t >= (long)N * Cout) return;
    int i = (int)(t / Cout), d = (int)(t % Cout);
    int k = kidx[i];
    const unsigned short* xr = xh + (long)i * Cin;
    const float* wr = w + (long)k * Cin * Cout + d;
    float acc = 0.0f;
    for (int c = 0; c < Cin; ++c) acc += h2f(xr[c]) * wr[(long)c * Cout];
    atomicAdd(out + (long)inv[i] * Cout + d, acc);
}

// ---------------------------------------------------------------------------
// stride-2 deconv: gather parent feature, apply W[kidx]; write second half of cat
// ---------------------------------------------------------------------------
__global__ void k_up(const unsigned short* __restrict__ xh, const int* __restrict__ inv,
                     const int* __restrict__ kidx, const float* __restrict__ w,
                     float* __restrict__ cat, int Nfine, int Cin, int Cout, int catC) {
    long t = (long)blockIdx.x * blockDim.x + threadIdx.x;
    if (t >= (long)Nfine * Cout) return;
    int i = (int)(t / Cout), d = (int)(t % Cout);
    int k = kidx[i];
    const unsigned short* xr = xh + (long)inv[i] * Cin;
    const float* wr = w + (long)k * Cin * Cout + d;
    float acc = 0.0f;
    for (int c = 0; c < Cin; ++c) acc += h2f(xr[c]) * wr[(long)c * Cout];
    cat[(long)i * catC + Cout + d] = acc;
}

// copy skip features into first half of concat buffer
__global__ void k_copycat(const float* __restrict__ feat, float* __restrict__ cat,
                          int N, int C, int catC) {
    long t = (long)blockIdx.x * blockDim.x + threadIdx.x;
    if (t >= (long)N * C) return;
    int i = (int)(t / C), c = (int)(t % C);
    cat[(long)i * catC + c] = feat[t];
}

// ---------------------------------------------------------------------------
// head: BN+ReLU (32 ch) + linear 32->7 + softmax
// ---------------------------------------------------------------------------
__global__ void k_head(const float* __restrict__ x, const float* __restrict__ stats,
                       const float* __restrict__ g, const float* __restrict__ b,
                       const float* __restrict__ wl, const float* __restrict__ bl,
                       float* __restrict__ out, int N) {
    int nidx = blockIdx.x * blockDim.x + threadIdx.x;
    if (nidx >= N) return;
    float z[7];
#pragma unroll
    for (int j = 0; j < 7; ++j) z[j] = bl[j];
    for (int c = 0; c < 32; ++c) {
        float v = (x[(long)nidx * 32 + c] - stats[c]) * stats[32 + c] * g[c] + b[c];
        v = v > 0.0f ? v : 0.0f;
#pragma unroll
        for (int j = 0; j < 7; ++j) z[j] += v * wl[c * 7 + j];
    }
    float mx = z[0];
#pragma unroll
    for (int j = 1; j < 7; ++j) mx = z[j] > mx ? z[j] : mx;
    float s = 0.0f;
#pragma unroll
    for (int j = 0; j < 7; ++j) { z[j] = __expf(z[j] - mx); s += z[j]; }
    float inv = 1.0f / s;
#pragma unroll
    for (int j = 0; j < 7; ++j) out[(long)nidx * 7 + j] = z[j] * inv;
}

// ---------------------------------------------------------------------------
// host orchestration
// ---------------------------------------------------------------------------
static inline unsigned gcount(long n, int bs) { return (unsigned)((n + bs - 1) / bs); }

extern "C" void kernel_launch(void* const* d_in, const int* in_sizes, int n_in,
                              void* d_out, int out_size, void* d_ws, size_t ws_size,
                              hipStream_t stream) {
    static const int C[5] = {32, 64, 96, 128, 160};
    int idx = 0;
    auto F = [&](void) { return (const float*)d_in[idx++]; };
    auto I = [&](void) { return (const int*)d_in[idx++]; };

    const float* feats = F();
    const float* w_in  = F();
    const float *bn_pre_g[5], *bn_pre_b[5];
    for (int l = 0; l < 5; ++l) { bn_pre_g[l] = F(); bn_pre_b[l] = F(); }
    const float* conv_pre[5];
    for (int l = 0; l < 5; ++l) conv_pre[l] = F();
    const float *bn_down_g[4], *bn_down_b[4];
    for (int l = 0; l < 4; ++l) { bn_down_g[l] = F(); bn_down_b[l] = F(); }
    const float* conv_down[4];
    for (int l = 0; l < 4; ++l) conv_down[l] = F();
    const float *bn_up_g[4], *bn_up_b[4];
    for (int l = 0; l < 4; ++l) { bn_up_g[l] = F(); bn_up_b[l] = F(); }
    const float* deconv[4];
    for (int l = 0; l < 4; ++l) deconv[l] = F();
    const float *bn_post_g[4], *bn_post_b[4];
    for (int l = 0; l < 4; ++l) { bn_post_g[l] = F(); bn_post_b[l] = F(); }
    const float* conv_post[4];
    for (int l = 0; l < 4; ++l) conv_post[l] = F();
    const float* bn_out_g = F();
    const float* bn_out_b = F();
    const float* w_lin = F();
    const float* b_lin = F();

    const int* nbrs[5]; int N[5];
    for (int l = 0; l < 5; ++l) { N[l] = in_sizes[idx] / 9; nbrs[l] = I(); }
    const int* invs[4];
    for (int l = 0; l < 4; ++l) invs[l] = I();
    const int* kidxs[4];
    for (int l = 0; l < 4; ++l) kidxs[l] = I();
    // nlev scalar arrays unused (levels recovered from rulebook sizes)

    size_t off = 0;
    auto take = [&](size_t bytes) -> void* {
        off = (off + 255) & ~(size_t)255;
        void* p = (char*)d_ws + off;
        off += bytes;
        return p;
    };
    float* feat[5];
    for (int l = 0; l < 5; ++l) feat[l] = (float*)take((size_t)N[l] * C[l] * 4);
    size_t maxcat = 0;
    for (int l = 0; l < 4; ++l) {
        size_t s = (size_t)N[l] * 2 * C[l];
        if (s > maxcat) maxcat = s;
    }
    size_t maxstage = maxcat;
    for (int l = 0; l < 5; ++l) {
        size_t s = (size_t)N[l] * C[l];
        if (s > maxstage) maxstage = s;
    }
    float* catb = (float*)take(maxcat * 4);
    unsigned short* hbuf = (unsigned short*)take(maxstage * 2);
    unsigned short* wpre[5];
    for (int l = 0; l < 5; ++l) wpre[l] = (unsigned short*)take((size_t)9 * C[l] * C[l] * 2);
    unsigned short* wpost[4];
    for (int l = 0; l < 4; ++l) wpost[l] = (unsigned short*)take((size_t)9 * 2 * C[l] * C[l] * 2);
    float* stats = (float*)take(2 * 320 * 4);
    (void)ws_size; (void)n_in;

    // weight conversion (fp32 -> f16, transposed for WMMA B operand)
    for (int l = 0; l < 5; ++l) {
        long tot = (long)9 * C[l] * C[l];
        k_cvt_wt<<<gcount(tot, 256), 256, 0, stream>>>(conv_pre[l], wpre[l], 9, C[l], C[l]);
    }
    for (int l = 0; l < 4; ++l) {
        long tot = (long)9 * 2 * C[l] * C[l];
        k_cvt_wt<<<gcount(tot, 256), 256, 0, stream>>>(conv_post[l], wpost[l], 9, 2 * C[l], C[l]);
    }

    // input conv 1 -> 32
    k_win<<<gcount((long)N[0] * 32, 256), 256, 0, stream>>>(feats, nbrs[0], w_in, feat[0], N[0]);

    // ---------------- encoder ----------------
    for (int l = 0; l < 5; ++l) {
        long tot = (long)N[l] * C[l];
        k_bn_stats<<<C[l], 256, 0, stream>>>(feat[l], stats, N[l], C[l]);
        k_bn_apply_h<<<gcount(tot, 256), 256, 0, stream>>>(feat[l], stats, bn_pre_g[l],
                                                           bn_pre_b[l], hbuf, tot, C[l]);
        dim3 grid(gcount(N[l], 64), C[l] / 16, 1);
        size_t shmem = (size_t)9 * 16 * C[l] * 2;
        k_smconv_wmma<<<grid, 128, shmem, stream>>>(hbuf, nbrs[l], wpre[l], feat[l],
                                                    N[l], C[l], C[l]);
        if (l < 4) {
            k_bn_stats<<<C[l], 256, 0, stream>>>(feat[l], stats, N[l], C[l]);
            k_bn_apply_h<<<gcount(tot, 256), 256, 0, stream>>>(feat[l], stats, bn_down_g[l],
                                                               bn_down_b[l], hbuf, tot, C[l]);
            long totn = (long)N[l + 1] * C[l + 1];
            k_zero_f32<<<gcount(totn, 256), 256, 0, stream>>>(feat[l + 1], totn);
            k_down<<<gcount((long)N[l] * C[l + 1], 256), 256, 0, stream>>>(
                hbuf, invs[l], kidxs[l], conv_down[l], feat[l + 1], N[l], C[l], C[l + 1]);
        }
    }

    // ---------------- decoder ----------------
    for (int l = 3; l >= 0; --l) {
        long totc = (long)N[l + 1] * C[l + 1];
        k_bn_stats<<<C[l + 1], 256, 0, stream>>>(feat[l + 1], stats, N[l + 1], C[l + 1]);
        k_bn_apply_h<<<gcount(totc, 256), 256, 0, stream>>>(feat[l + 1], stats, bn_up_g[l],
                                                            bn_up_b[l], hbuf, totc, C[l + 1]);
        long totf = (long)N[l] * C[l];
        k_copycat<<<gcount(totf, 256), 256, 0, stream>>>(feat[l], catb, N[l], C[l], 2 * C[l]);
        k_up<<<gcount(totf, 256), 256, 0, stream>>>(hbuf, invs[l], kidxs[l], deconv[l],
                                                    catb, N[l], C[l + 1], C[l], 2 * C[l]);
        long tot2 = (long)N[l] * 2 * C[l];
        k_bn_stats<<<2 * C[l], 256, 0, stream>>>(catb, stats, N[l], 2 * C[l]);
        k_bn_apply_h<<<gcount(tot2, 256), 256, 0, stream>>>(catb, stats, bn_post_g[l],
                                                            bn_post_b[l], hbuf, tot2, 2 * C[l]);
        dim3 grid(gcount(N[l], 64), C[l] / 16, 1);
        size_t shmem = (size_t)9 * 16 * (2 * C[l]) * 2;
        k_smconv_wmma<<<grid, 128, shmem, stream>>>(hbuf, nbrs[l], wpost[l], feat[l],
                                                    N[l], 2 * C[l], C[l]);
    }

    // ---------------- head ----------------
    k_bn_stats<<<32, 256, 0, stream>>>(feat[0], stats, N[0], 32);
    k_head<<<gcount(N[0], 128), 128, 0, stream>>>(feat[0], stats, bn_out_g, bn_out_b,
                                                  w_lin, b_lin, (float*)d_out, N[0]);
    (void)out_size;
}
